// _MatrixDecomposition2DBase_901943132540
// MI455X (gfx1250) — matrix operand order repaired and checked by reading, not compiled, not run on hardware
//
#include <hip/hip_runtime.h>
#include <hip/hip_bf16.h>

// ---------------------------------------------------------------------------
// NMF (MatrixDecomposition2D) for MI455X / gfx1250.
// All heavy contractions use v_wmma_f32_16x16x32_f16 (wave32 WMMA).
// ---------------------------------------------------------------------------

typedef __attribute__((ext_vector_type(16))) _Float16 v16h;
typedef __attribute__((ext_vector_type(8)))  _Float16 v8h;
typedef __attribute__((ext_vector_type(8)))  float    v8f;

union ABu { v16h v; v8h h[2]; };

#define EPS_ 1e-6f
#define B_  16
#define D_  512
#define N_  9216
#define R_  64

// ---- WMMA fragment loaders --------------------------------------------------
// A (16x32 f16): lane m=lane&15 holds row m. lanes<16: K0-7 (v0-3), K16-23 (v4-7)
//                lanes>=16: K8-15, K24-31. Source: row-major (M x K), lda elems.
__device__ __forceinline__ v16h load_A(const _Float16* base, int lda, int lane) {
  int m = lane & 15, sel = lane >> 4;
  const _Float16* p = base + (size_t)m * lda + sel * 8;
  ABu u;
  u.h[0] = *(const v8h*)p;          // 16 bytes
  u.h[1] = *(const v8h*)(p + 16);   // +32 bytes
  return u.v;
}

// B (32x16 f16): lane holds column c=lane&15; lanes<16: K0-7, K16-23; lanes>=16: K8-15, K24-31.
// Source: "BT" storage = 16 columns stored as contiguous-K rows, stride ldb.
__device__ __forceinline__ v16h load_B(const _Float16* baseT, int ldb, int lane) {
  int c = lane & 15, sel = lane >> 4;
  const _Float16* q = baseT + (size_t)c * ldb + sel * 8;
  ABu u;
  u.h[0] = *(const v8h*)q;
  u.h[1] = *(const v8h*)(q + 16);
  return u.v;
}

__device__ __forceinline__ v8f wmma_f16(v16h a, v16h b, v8f c) {
  return __builtin_amdgcn_wmma_f32_16x16x32_f16(false, a, false, b, (short)0, c,
                                                false, false);
}

// ---- prep: bases master + f16 + f16-transposed ------------------------------
__global__ void k_prep_bases(const float* __restrict__ bases_in,
                             float* __restrict__ bases,
                             _Float16* __restrict__ bases_h,
                             _Float16* __restrict__ basesT_h) {
  int idx = blockIdx.x * blockDim.x + threadIdx.x;
  if (idx >= B_ * D_ * R_) return;
  int r = idx % R_;
  int d = (idx / R_) % D_;
  int b = idx / (R_ * D_);
  float v = bases_in[idx];
  bases[idx] = v;
  bases_h[idx] = (_Float16)v;
  basesT_h[((size_t)b * R_ + r) * D_ + d] = (_Float16)v;
}

// ---- convert x -> f16 (D,N) and f16 (N,D) via LDS transpose -----------------
__global__ void k_convert_x(const float* __restrict__ x,
                            _Float16* __restrict__ xh,
                            _Float16* __restrict__ xth) {
  __shared__ _Float16 tile[32][33];
  int b = blockIdx.z;
  int d0 = blockIdx.y * 32, n0 = blockIdx.x * 32;
  const float* xb = x + (size_t)b * D_ * N_;
  _Float16* xhb = xh + (size_t)b * D_ * N_;
  _Float16* xtb = xth + (size_t)b * N_ * D_;
  int t = threadIdx.x;
  int tn = t & 31;
  int td = t >> 5;  // 0..7
  for (int i = 0; i < 4; ++i) {
    int d = td + i * 8;
    _Float16 h = (_Float16)xb[(size_t)(d0 + d) * N_ + n0 + tn];
    xhb[(size_t)(d0 + d) * N_ + n0 + tn] = h;
    tile[d][tn] = h;
  }
  __syncthreads();
  for (int i = 0; i < 4; ++i) {
    int nr = td + i * 8;
    xtb[(size_t)(n0 + nr) * D_ + d0 + tn] = tile[tn][nr];
  }
}

// ---- init coef = softmax(100 * xT @ bases) ----------------------------------
__global__ void k_init_coef(const _Float16* __restrict__ xth,
                            const _Float16* __restrict__ basesT_h,
                            float* __restrict__ coef,
                            _Float16* __restrict__ coef_h,
                            _Float16* __restrict__ coefT_h) {
  __shared__ float sNum[64][65];
  int b = blockIdx.y;
  int n0 = blockIdx.x * 64;
  const _Float16* A = xth + (size_t)b * N_ * D_;       // (N, D)
  const _Float16* BT = basesT_h + (size_t)b * R_ * D_; // (R, D)
  int t = threadIdx.x, lane = t & 31, wave = t >> 5;

  for (int ti = 0; ti < 2; ++ti) {
    int tile = wave * 2 + ti;
    int tr = tile >> 2, tc = tile & 3;
    const _Float16* Ab = A + (size_t)(n0 + tr * 16) * D_;
    const _Float16* Bb = BT + (size_t)(tc * 16) * D_;
    v8f acc = {};
    for (int k = 0; k < D_; k += 32)
      acc = wmma_f16(load_A(Ab + k, D_, lane), load_B(Bb + k, D_, lane), acc);
    int c = lane & 15, sel = lane >> 4;
    for (int j = 0; j < 8; ++j) sNum[tr * 16 + sel * 8 + j][tc * 16 + c] = acc[j];
  }
  __syncthreads();

  if (t < 64) {
    int row = t;
    float mx = -1e30f;
    for (int r = 0; r < 64; ++r) mx = fmaxf(mx, sNum[row][r]);
    float sum = 0.f;
    for (int r = 0; r < 64; ++r) sum += __expf(100.0f * (sNum[row][r] - mx));
    float inv = 1.0f / sum;
    size_t base = ((size_t)b * N_ + n0 + row) * (size_t)R_;
    for (int r = 0; r < 64; ++r) {
      float cv = __expf(100.0f * (sNum[row][r] - mx)) * inv;
      coef[base + r] = cv;
      coef_h[base + r] = (_Float16)cv;
      coefT_h[((size_t)b * R_ + r) * N_ + n0 + row] = (_Float16)cv;
    }
  }
}

// ---- gram: G = MT * MT^T, MT is (R x K) f16 row-major per batch -------------
__global__ void k_gram(const _Float16* __restrict__ MT, float* __restrict__ G,
                       int K) {
  int b = blockIdx.x;
  const _Float16* M = MT + (size_t)b * R_ * K;
  float* Gb = G + (size_t)b * R_ * R_;
  int t = threadIdx.x, lane = t & 31, wave = t >> 5;
  for (int ti = 0; ti < 2; ++ti) {
    int tile = wave * 2 + ti;
    int tr = tile >> 2, tc = tile & 3;
    const _Float16* Ab = M + (size_t)(tr * 16) * K;
    const _Float16* Bb = M + (size_t)(tc * 16) * K;
    v8f acc = {};
    for (int k = 0; k < K; k += 32)
      acc = wmma_f16(load_A(Ab + k, K, lane), load_B(Bb + k, K, lane), acc);
    int c = lane & 15, sel = lane >> 4;
    for (int j = 0; j < 8; ++j) Gb[(tr * 16 + sel * 8 + j) * R_ + tc * 16 + c] = acc[j];
  }
}

// ---- coef multiplicative update ---------------------------------------------
// numer = xT@bases (WMMA); denom = coef@BtB (+eps); coef *= numer/denom
__global__ void k_coef_step(const _Float16* __restrict__ xth,
                            const _Float16* __restrict__ basesT_h,
                            const float* __restrict__ BtB,
                            float* __restrict__ coef,
                            _Float16* __restrict__ coef_h,
                            _Float16* __restrict__ coefT_h) {
  __shared__ float sNum[64][65];
  __shared__ float sG[64][65];
  __shared__ float sC[64][65];
  int b = blockIdx.y;
  int n0 = blockIdx.x * 64;
  int t = threadIdx.x, lane = t & 31, wave = t >> 5;

  const float* Gb = BtB + (size_t)b * R_ * R_;
  const float* Cb = coef + ((size_t)b * N_ + n0) * (size_t)R_;
  for (int i = t; i < 4096; i += 256) {
    sG[i >> 6][i & 63] = Gb[i];
    sC[i >> 6][i & 63] = Cb[i];
  }

  const _Float16* A = xth + (size_t)b * N_ * D_;
  const _Float16* BT = basesT_h + (size_t)b * R_ * D_;
  for (int ti = 0; ti < 2; ++ti) {
    int tile = wave * 2 + ti;
    int tr = tile >> 2, tc = tile & 3;
    const _Float16* Ab = A + (size_t)(n0 + tr * 16) * D_;
    const _Float16* Bb = BT + (size_t)(tc * 16) * D_;
    v8f acc = {};
    for (int k = 0; k < D_; k += 32)
      acc = wmma_f16(load_A(Ab + k, D_, lane), load_B(Bb + k, D_, lane), acc);
    int c = lane & 15, sel = lane >> 4;
    for (int j = 0; j < 8; ++j) sNum[tr * 16 + sel * 8 + j][tc * 16 + c] = acc[j];
  }
  __syncthreads();

  int row = t & 63, q = t >> 6;  // q selects 16 r-values
  float denom[16];
  for (int j = 0; j < 16; ++j) denom[j] = 0.f;
  for (int s = 0; s < 64; ++s) {
    float cs = sC[row][s];
    for (int j = 0; j < 16; ++j) denom[j] += cs * sG[s][q * 16 + j];
  }
  size_t cbase = ((size_t)b * N_ + n0 + row) * (size_t)R_;
  for (int j = 0; j < 16; ++j) {
    int r = q * 16 + j;
    float nc = sC[row][r] * sNum[row][r] / (denom[j] + EPS_);
    coef[cbase + r] = nc;
    coef_h[cbase + r] = (_Float16)nc;
    coefT_h[((size_t)b * R_ + r) * N_ + n0 + row] = (_Float16)nc;
  }
}

// ---- bases multiplicative update --------------------------------------------
// numer = x@coef (WMMA, K=N); denom = bases@CtC (+eps); bases *= numer/denom
__global__ void k_bases_step(const _Float16* __restrict__ xh,
                             const _Float16* __restrict__ coefT_h,
                             const float* __restrict__ CtC,
                             float* __restrict__ bases,
                             _Float16* __restrict__ bases_h,
                             _Float16* __restrict__ basesT_h) {
  __shared__ float sNum[64][65];
  __shared__ float sG[64][65];
  __shared__ float sB[64][65];
  int b = blockIdx.y;
  int d0 = blockIdx.x * 64;
  int t = threadIdx.x, lane = t & 31, wave = t >> 5;

  const float* Gb = CtC + (size_t)b * R_ * R_;
  const float* Bbm = bases + ((size_t)b * D_ + d0) * (size_t)R_;
  for (int i = t; i < 4096; i += 256) {
    sG[i >> 6][i & 63] = Gb[i];
    sB[i >> 6][i & 63] = Bbm[i];
  }

  const _Float16* A = xh + (size_t)b * D_ * N_;        // (D, N)
  const _Float16* BT = coefT_h + (size_t)b * R_ * N_;  // (R, N)
  for (int ti = 0; ti < 2; ++ti) {
    int tile = wave * 2 + ti;
    int tr = tile >> 2, tc = tile & 3;
    const _Float16* Ab = A + (size_t)(d0 + tr * 16) * N_;
    const _Float16* Bb = BT + (size_t)(tc * 16) * N_;
    v8f acc = {};
    for (int k = 0; k < N_; k += 32)
      acc = wmma_f16(load_A(Ab + k, N_, lane), load_B(Bb + k, N_, lane), acc);
    int c = lane & 15, sel = lane >> 4;
    for (int j = 0; j < 8; ++j) sNum[tr * 16 + sel * 8 + j][tc * 16 + c] = acc[j];
  }
  __syncthreads();

  int row = t & 63, q = t >> 6;
  float denom[16];
  for (int j = 0; j < 16; ++j) denom[j] = 0.f;
  for (int s = 0; s < 64; ++s) {
    float bs = sB[row][s];
    for (int j = 0; j < 16; ++j) denom[j] += bs * sG[s][q * 16 + j];
  }
  size_t bbase = ((size_t)b * D_ + d0 + row) * (size_t)R_;
  for (int j = 0; j < 16; ++j) {
    int r = q * 16 + j;
    float nb = sB[row][r] * sNum[row][r] / (denom[j] + EPS_);
    bases[bbase + r] = nb;
    bases_h[bbase + r] = (_Float16)nb;
    basesT_h[((size_t)b * R_ + r) * D_ + d0 + row] = (_Float16)nb;
  }
}

// ---- reconstruction: out = bases @ coef^T (K = R = 64) ----------------------
__global__ void k_recon(const _Float16* __restrict__ bases_h,
                        const _Float16* __restrict__ coef_h,
                        float* __restrict__ out) {
  int b = blockIdx.z;
  int d0 = blockIdx.y * 64, n0 = blockIdx.x * 64;
  const _Float16* A = bases_h + ((size_t)b * D_ + d0) * (size_t)R_; // (D,R)
  const _Float16* BT = coef_h + ((size_t)b * N_ + n0) * (size_t)R_; // (N,R): col n contiguous in r
  float* O = out + (size_t)b * D_ * N_;
  int t = threadIdx.x, lane = t & 31, wave = t >> 5;
  for (int ti = 0; ti < 2; ++ti) {
    int tile = wave * 2 + ti;
    int tr = tile >> 2, tc = tile & 3;  // tr: d-sub, tc: n-sub
    const _Float16* Ab = A + (size_t)(tr * 16) * R_;
    const _Float16* Bb = BT + (size_t)(tc * 16) * R_;
    v8f acc = {};
    for (int k = 0; k < R_; k += 32)
      acc = wmma_f16(load_A(Ab + k, R_, lane), load_B(Bb + k, R_, lane), acc);
    int c = lane & 15, sel = lane >> 4;
    for (int j = 0; j < 8; ++j)
      O[(size_t)(d0 + tr * 16 + sel * 8 + j) * N_ + n0 + tc * 16 + c] = acc[j];
  }
}

// ---------------------------------------------------------------------------
extern "C" void kernel_launch(void* const* d_in, const int* in_sizes, int n_in,
                              void* d_out, int out_size, void* d_ws, size_t ws_size,
                              hipStream_t stream) {
  (void)in_sizes; (void)n_in; (void)out_size; (void)ws_size;
  const float* x = (const float*)d_in[0];
  const float* bases_in = (const float*)d_in[1];
  float* out = (float*)d_out;

  char* ws = (char*)d_ws;
  float*    coef     = (float*)ws;               ws += (size_t)B_ * N_ * R_ * 4;
  float*    bases    = (float*)ws;               ws += (size_t)B_ * D_ * R_ * 4;
  float*    BtB      = (float*)ws;               ws += (size_t)B_ * R_ * R_ * 4;
  float*    CtC      = (float*)ws;               ws += (size_t)B_ * R_ * R_ * 4;
  _Float16* xh       = (_Float16*)ws;            ws += (size_t)B_ * D_ * N_ * 2;
  _Float16* xth      = (_Float16*)ws;            ws += (size_t)B_ * N_ * D_ * 2;
  _Float16* coef_h   = (_Float16*)ws;            ws += (size_t)B_ * N_ * R_ * 2;
  _Float16* coefT_h  = (_Float16*)ws;            ws += (size_t)B_ * R_ * N_ * 2;
  _Float16* bases_h  = (_Float16*)ws;            ws += (size_t)B_ * D_ * R_ * 2;
  _Float16* basesT_h = (_Float16*)ws;            ws += (size_t)B_ * R_ * D_ * 2;

  k_prep_bases<<<(B_ * D_ * R_ + 255) / 256, 256, 0, stream>>>(bases_in, bases,
                                                               bases_h, basesT_h);
  k_convert_x<<<dim3(N_ / 32, D_ / 32, B_), 256, 0, stream>>>(x, xh, xth);
  k_init_coef<<<dim3(N_ / 64, B_), 256, 0, stream>>>(xth, basesT_h, coef, coef_h,
                                                     coefT_h);
  for (int s = 0; s < 7; ++s) {
    k_gram<<<B_, 256, 0, stream>>>(basesT_h, BtB, D_);
    k_coef_step<<<dim3(N_ / 64, B_), 256, 0, stream>>>(xth, basesT_h, BtB, coef,
                                                       coef_h, coefT_h);
    k_gram<<<B_, 256, 0, stream>>>(coefT_h, CtC, N_);
    k_bases_step<<<dim3(D_ / 64, B_), 256, 0, stream>>>(xh, coefT_h, CtC, bases,
                                                        bases_h, basesT_h);
  }
  // final compute_coef
  k_gram<<<B_, 256, 0, stream>>>(basesT_h, BtB, D_);
  k_coef_step<<<dim3(N_ / 64, B_), 256, 0, stream>>>(xth, basesT_h, BtB, coef,
                                                     coef_h, coefT_h);
  // reconstruction -> d_out
  k_recon<<<dim3(N_ / 64, D_ / 64, B_), 256, 0, stream>>>(bases_h, coef_h, out);
}
